// MultiHeadAttentionLayer_21715354648981
// MI455X (gfx1250) — matrix operand order repaired and checked by reading, not compiled, not run on hardware
//
#include <hip/hip_runtime.h>
#include <hip/hip_bf16.h>
#include <cstdint>

// ---------------------------------------------------------------------------
// MHA (B=4,S=2048,D=1024,H=16,dk=64) for MI455X / gfx1250, wave32 + WMMA bf16.
//   proj   = x @ Wqkv + bqkv                      (WMMA GEMM, bf16 out)
//   per head h: QKV = contiguous [2048,64] slice  (raw view quirk!)
//   flash attention (online softmax, fp32 stats)  (WMMA)
//   out    = merged @ Wo + bo                     (WMMA GEMM, fp32 out)
// Tile staging uses the Tensor Data Mover (tensor_load_to_lds + TENSORcnt)
// when the toolchain exposes it; otherwise cooperative VGPR staging.
// ---------------------------------------------------------------------------

typedef __attribute__((ext_vector_type(16))) __bf16 v16bf;
typedef __attribute__((ext_vector_type(8)))  __bf16 v8bf;
typedef __attribute__((ext_vector_type(8)))  float  v8f;

#define SHUF16(lo, hi) \
  __builtin_shufflevector(lo, hi, 0,1,2,3,4,5,6,7,8,9,10,11,12,13,14,15)

static __device__ __forceinline__ v8f wmma_bf16(v16bf a, v16bf b, v8f c) {
  // D = A(16x32 bf16) * B(32x16 bf16) + C(16x16 f32)
  return __builtin_amdgcn_wmma_f32_16x16x32_bf16(
      /*neg_a=*/false, a, /*neg_b=*/false, b,
      /*c_mod=*/(short)0, c, /*reuse_a=*/false, /*reuse_b=*/false);
}

#define DM    1024
#define SEQ   2048
#define BATCH 4
#define HEADS 16
#define DK    64
#define MROWS (BATCH * SEQ)   // 8192

// ---------------------------------------------------------------------------
// Tensor Data Mover (gfx1250). D# packing per CDNA5 ISA ch.8:
//  group0: [1:0]=count  [63:32]=lds_addr  [120:64]=global_addr  [127:126]=type(2)
//  group1: [17:16]=data_size(1=2B) [79:48]=tensor_dim0 [111:80]=tensor_dim1
//          [127:112]=tile_dim0 [143:128]=tile_dim1 [207:160]=tensor_dim0_stride
// 2-D tiles only -> descriptor groups 2/3 zero-filled.
// ---------------------------------------------------------------------------
#if __has_builtin(__builtin_amdgcn_tensor_load_to_lds) && \
    __has_builtin(__builtin_amdgcn_s_wait_tensorcnt)
#define HAVE_TDM 1
#else
#define HAVE_TDM 0
#endif

#if HAVE_TDM
typedef __attribute__((ext_vector_type(4))) unsigned tdm_v4u;
typedef __attribute__((ext_vector_type(4))) int      tdm_v4i;
typedef __attribute__((ext_vector_type(8))) int      tdm_v8i;

static __device__ __forceinline__ void tdm_load_2d(
    unsigned lds_byte_addr, const void* gaddr,
    unsigned tensor_d0, unsigned tensor_d1,
    unsigned tile_d0, unsigned tile_d1, unsigned stride_d0_elems) {
  unsigned long long ga = (unsigned long long)gaddr;
  tdm_v4u g0;
  g0.x = 1u;                                            // count=1 (valid), user
  g0.y = lds_byte_addr;                                 // LDS byte address
  g0.z = (unsigned)(ga & 0xFFFFFFFFu);                  // global_addr[31:0]
  g0.w = (unsigned)((ga >> 32) & 0x01FFFFFFu) | (2u << 30);  // [56:32] | type=2
  tdm_v8i g1;
  g1[0] = (int)(1u << 16);                              // data_size = 2 bytes
  g1[1] = (int)((tensor_d0 & 0xFFFFu) << 16);           // tensor_dim0[15:0]
  g1[2] = (int)(((tensor_d0 >> 16) & 0xFFFFu) | ((tensor_d1 & 0xFFFFu) << 16));
  g1[3] = (int)(((tensor_d1 >> 16) & 0xFFFFu) | ((tile_d0 & 0xFFFFu) << 16));
  g1[4] = (int)(tile_d1 & 0xFFFFu);                     // tile_dim1, tile_dim2=0
  g1[5] = (int)stride_d0_elems;                         // tensor_dim0_stride lo32
  g1[6] = 0;                                            // stride0 hi | stride1 lo
  g1[7] = 0;                                            // stride1 hi
  tdm_v4i gz = {0, 0, 0, 0};
#if __clang_major__ >= 23
  tdm_v8i gz8 = {0, 0, 0, 0, 0, 0, 0, 0};
  __builtin_amdgcn_tensor_load_to_lds(g0, g1, gz, gz, gz8, 0);
#else
  __builtin_amdgcn_tensor_load_to_lds(g0, g1, gz, gz, 0);
#endif
}
#endif  // HAVE_TDM

// ---------------------------------------------------------------------------
// fp32 -> bf16 (4 elems / thread)
// ---------------------------------------------------------------------------
__global__ void cvt_f32_to_bf16(const float* __restrict__ src,
                                __bf16* __restrict__ dst, int n) {
  int i = (blockIdx.x * blockDim.x + threadIdx.x) * 4;
  if (i + 3 < n) {
    float4 v = *(const float4*)(src + i);
    dst[i + 0] = (__bf16)v.x;
    dst[i + 1] = (__bf16)v.y;
    dst[i + 2] = (__bf16)v.z;
    dst[i + 3] = (__bf16)v.w;
  }
}

// Wt[n][k] = (bf16)W[k][n]   (D x D)  -> GEMM B operand is K-contiguous
__global__ void cvt_transpose_bf16(const float* __restrict__ W,
                                   __bf16* __restrict__ Wt, int D) {
  int idx = blockIdx.x * blockDim.x + threadIdx.x;
  if (idx < D * D) {
    int k = idx / D, n = idx % D;       // coalesced read over n
    Wt[(size_t)n * D + k] = (__bf16)W[(size_t)k * D + n];
  }
}

// ---------------------------------------------------------------------------
// WMMA GEMM: C[M,N] = A[M,K](bf16,row) * Bt[N,K](bf16,row == B^T) + bias[N]
// 256 thr = 8 waves; WG tile 128x128; wave tile 64(M) x 32(N); K-step 32.
// ---------------------------------------------------------------------------
#define TM 128
#define TN 128
#define TK 32

template <bool OUT_BF16>
__global__ __launch_bounds__(256)
void gemm_wmma_bf16(const __bf16* __restrict__ A,
                    const __bf16* __restrict__ Bt,
                    const float*  __restrict__ bias,
                    float*  __restrict__ outF,
                    __bf16* __restrict__ outB,
                    int M, int N, int K) {
  __shared__ __align__(32) __bf16 As[TM][TK];   // 8 KB
  __shared__ __align__(32) __bf16 Bs[TN][TK];   // 8 KB (n-major == B^T tile)

  const int tid     = threadIdx.x;
  const int lane    = tid & 31;
  const int wave    = tid >> 5;          // 0..7
  const int waveM   = wave >> 2;         // 0..1  (64 rows each)
  const int waveN   = wave & 3;          // 0..3  (32 cols each)
  const int l15     = lane & 15;
  const int halfsel = (lane < 16) ? 0 : 1;
  const int tileM   = blockIdx.x * TM;
  const int tileN   = blockIdx.y * TN;

  v8f acc[4][2] = {};                    // 4 M-blocks x 2 N-blocks, 16x16 f32 tiles

  for (int kk = 0; kk < K; kk += TK) {
#if HAVE_TDM
    // -------- TDM: DMA both tiles into LDS, wave 0 drives TENSORcnt --------
    if (wave == 0) {
      tdm_load_2d((unsigned)(unsigned long long)(void*)&As[0][0],
                  A + (size_t)tileM * K + kk,
                  (unsigned)K, (unsigned)M, TK, TM, (unsigned)K);
      tdm_load_2d((unsigned)(unsigned long long)(void*)&Bs[0][0],
                  Bt + (size_t)tileN * K + kk,
                  (unsigned)K, (unsigned)N, TK, TN, (unsigned)K);
      __builtin_amdgcn_s_wait_tensorcnt(0);
    }
    __syncthreads();
#else
    // -------- cooperative staging: 2 threads/row, 16 halfs each ------------
    {
      const int srow = tid >> 1, sseg = tid & 1;
      const uint4* ga = (const uint4*)(A  + (size_t)(tileM + srow) * K + kk + sseg * 16);
      const uint4* gb = (const uint4*)(Bt + (size_t)(tileN + srow) * K + kk + sseg * 16);
      uint4 a0 = ga[0], a1 = ga[1];
      uint4 b0 = gb[0], b1 = gb[1];
      uint4* sa = (uint4*)&As[srow][sseg * 16];
      uint4* sb = (uint4*)&Bs[srow][sseg * 16];
      sa[0] = a0; sa[1] = a1;
      sb[0] = b0; sb[1] = b1;
      if (kk + TK < K) {                 // gfx1250 global_prefetch_b8
        __builtin_prefetch(A  + (size_t)(tileM + srow) * K + kk + TK + sseg * 16, 0, 1);
        __builtin_prefetch(Bt + (size_t)(tileN + srow) * K + kk + TK + sseg * 16, 0, 1);
      }
    }
    __syncthreads();
#endif

    // -------- fragments ----------------------------------------------------
    // A layout (16x32 bf16): lane<16 holds row l15, K=0..15 contiguous;
    //                        lane>=16 same row,    K=16..31.
    const int kc = halfsel * 8;
    v16bf afr[4];
#pragma unroll
    for (int mb = 0; mb < 4; ++mb) {
      int r = waveM * 64 + mb * 16 + l15;
      afr[mb] = *(const v16bf*)&As[r][halfsel * 16];
    }
    // B layout (32x16 bf16): lane<16 holds col l15, K=0..15 contiguous;
    //                        lane>=16 holds col l15, K=16..31.
    v16bf bfr[2];
#pragma unroll
    for (int nb = 0; nb < 2; ++nb) {
      int c = waveN * 32 + nb * 16 + l15;
      bfr[nb] = *(const v16bf*)&Bs[c][halfsel * 16];
    }
#pragma unroll
    for (int mb = 0; mb < 4; ++mb)
#pragma unroll
      for (int nb = 0; nb < 2; ++nb)
        acc[mb][nb] = wmma_bf16(afr[mb], bfr[nb], acc[mb][nb]);
    __syncthreads();
  }

  // -------- epilogue: C layout: VGPR r -> row (r + 8*halfsel), col l15 ------
#pragma unroll
  for (int nb = 0; nb < 2; ++nb) {
    int col  = tileN + waveN * 32 + nb * 16 + l15;
    float bv = bias[col];
#pragma unroll
    for (int mb = 0; mb < 4; ++mb) {
#pragma unroll
      for (int r = 0; r < 8; ++r) {
        int row = tileM + waveM * 64 + mb * 16 + r + (halfsel ? 8 : 0);
        float v = acc[mb][nb][r] + bv;
        if (OUT_BF16) outB[(size_t)row * N + col] = (__bf16)v;
        else          outF[(size_t)row * N + col] = v;
      }
    }
  }
}

// ---------------------------------------------------------------------------
// Flash attention per head-slice. Head h of batch b is the CONTIGUOUS slice
// proj + (b*H + h)*S*dk, viewed as row-major [2048 x 64] (raw-view quirk).
// WG = 128 thr (4 waves); wave owns a 16-row Q block; WG covers 64 Q rows.
// Loop over 32-key tiles: S_t = Q Kt^T, online softmax (f32), F += P V.
// ---------------------------------------------------------------------------
__global__ __launch_bounds__(128)
void attn_wmma(const __bf16* __restrict__ proj,   // [B,S,D] bf16
               __bf16* __restrict__ merged) {     // [B,S,D] bf16
  __shared__ __align__(32) __bf16 Ks[32][DK];     // 4 KB key tile, row-major
  __shared__ __align__(32) __bf16 Vt[DK][32];     // 4 KB transposed (for P@V)
  __shared__ __align__(32) __bf16 Ps[4][16][32];  // 4 KB per-wave P staging

  const int tid     = threadIdx.x;
  const int lane    = tid & 31;
  const int wave    = tid >> 5;                   // 0..3
  const int l15     = lane & 15;
  const int halfsel = (lane < 16) ? 0 : 1;
  const int kc      = halfsel * 8;

  const int h = blockIdx.y, b = blockIdx.z;
  const __bf16* head = proj + ((size_t)b * HEADS + h) * (size_t)(SEQ * DK);
  const int qBase = blockIdx.x * 64;

  // ---- persistent Q fragments: rows qBase + wave*16 + l15, dk 0..31/32..63
  const int qrow = qBase + wave * 16 + l15;
  v16bf qf[2];
#pragma unroll
  for (int ks = 0; ks < 2; ++ks) {
    qf[ks] = *(const v16bf*)(head + (size_t)qrow * DK + ks * 32 + halfsel * 16);
  }

  float m[8], l[8];
  v8f   fa[4] = {};
#pragma unroll
  for (int r = 0; r < 8; ++r) { m[r] = -1.0e30f; l[r] = 0.0f; }

  const int krow = tid >> 2, kseg = tid & 3;      // 32 rows x 4 segs of 16 halfs

  for (int kt = 0; kt < SEQ / 32; ++kt) {
#if HAVE_TDM
    // -------- TDM: key tile is one contiguous 4 KB run -> 1-D descriptor ---
    if (wave == 0) {
      tdm_load_2d((unsigned)(unsigned long long)(void*)&Ks[0][0],
                  head + (size_t)(kt * 32) * DK,
                  /*tensor_d0=*/32 * DK, /*tensor_d1=*/1,
                  /*tile_d0=*/32 * DK, /*tile_d1=*/1, /*stride_d0=*/32 * DK);
      __builtin_amdgcn_s_wait_tensorcnt(0);
    }
    __syncthreads();
    // build Vt (transpose) out of the DMA'd tile
    {
      union { uint4 u[2]; __bf16 hh[16]; } tmp;
      tmp.u[0] = *(const uint4*)&Ks[krow][kseg * 16];
      tmp.u[1] = *((const uint4*)&Ks[krow][kseg * 16] + 1);
#pragma unroll
      for (int i = 0; i < 16; ++i) Vt[kseg * 16 + i][krow] = tmp.hh[i];
    }
#else
    // -------- cooperative staging + register transpose ---------------------
    {
      const uint4* g = (const uint4*)(head + (size_t)(kt * 32 + krow) * DK + kseg * 16);
      uint4 d0 = g[0], d1 = g[1];
      uint4* s = (uint4*)&Ks[krow][kseg * 16];
      s[0] = d0; s[1] = d1;
      union { uint4 u[2]; __bf16 hh[16]; } tmp;
      tmp.u[0] = d0; tmp.u[1] = d1;
#pragma unroll
      for (int i = 0; i < 16; ++i) Vt[kseg * 16 + i][krow] = tmp.hh[i];
      if (kt + 1 < SEQ / 32)
        __builtin_prefetch(head + (size_t)((kt + 1) * 32 + krow) * DK + kseg * 16, 0, 1);
    }
    __syncthreads();
#endif

    // -------- scores S = Q Kt^T * 0.125 ------------------------------------
    v8f sc[2];
#pragma unroll
    for (int nb = 0; nb < 2; ++nb) {
      int key = nb * 16 + l15;                    // B-frag: lane holds K row `key`
      v16bf b0 = *(const v16bf*)&Ks[key][0  + halfsel * 16];   // dk 0..31
      v16bf b1 = *(const v16bf*)&Ks[key][32 + halfsel * 16];   // dk 32..63
      v8f a = {};
      a = wmma_bf16(qf[0], b0, a);
      a = wmma_bf16(qf[1], b1, a);
      sc[nb] = a;
    }

    // -------- online softmax (row = r + 8*halfsel, spread over 16 lanes) ---
#pragma unroll
    for (int r = 0; r < 8; ++r) {
      float s0 = sc[0][r] * 0.125f;
      float s1 = sc[1][r] * 0.125f;
      float tm = fmaxf(s0, s1);
      tm = fmaxf(tm, __shfl_xor(tm, 1));
      tm = fmaxf(tm, __shfl_xor(tm, 2));
      tm = fmaxf(tm, __shfl_xor(tm, 4));
      tm = fmaxf(tm, __shfl_xor(tm, 8));
      float mn    = fmaxf(m[r], tm);
      float alpha = __expf(m[r] - mn);
      float p0    = __expf(s0 - mn);
      float p1    = __expf(s1 - mn);
      float rs = p0 + p1;
      rs += __shfl_xor(rs, 1);
      rs += __shfl_xor(rs, 2);
      rs += __shfl_xor(rs, 4);
      rs += __shfl_xor(rs, 8);
      l[r] = l[r] * alpha + rs;
      m[r] = mn;
#pragma unroll
      for (int db = 0; db < 4; ++db) fa[db][r] *= alpha;
      // C-layout -> LDS (so we can re-read in A-layout)
      int prow = r + (halfsel ? 8 : 0);
      Ps[wave][prow][l15]      = (__bf16)p0;
      Ps[wave][prow][16 + l15] = (__bf16)p1;
    }
    __syncthreads();   // publishes Ps and Vt

    // -------- F += P V  (A-frag of P, B-frags from Vt) ---------------------
    v16bf pf = *(const v16bf*)&Ps[wave][l15][halfsel * 16];
#pragma unroll
    for (int db = 0; db < 4; ++db) {
      int n = db * 16 + l15;                      // output dk column
      v16bf vf = *(const v16bf*)&Vt[n][halfsel * 16];
      fa[db] = wmma_bf16(pf, vf, fa[db]);
    }
    __syncthreads();   // all reads of Ks/Vt done before next restage
  }

  // -------- normalize + write merged[b, s, h*64 + dk] ----------------------
  const size_t outBase = (size_t)b * SEQ * DM + (size_t)h * DK;
#pragma unroll
  for (int db = 0; db < 4; ++db) {
    int col = db * 16 + l15;
#pragma unroll
    for (int r = 0; r < 8; ++r) {
      int row  = qBase + wave * 16 + r + (halfsel ? 8 : 0);
      float v  = fa[db][r] / l[r];
      merged[outBase + (size_t)row * DM + col] = (__bf16)v;
    }
  }
}

// ---------------------------------------------------------------------------
// Host-side launch. Inputs: x, Wqkv, bqkv, Wo, bo (all fp32). Output fp32.
// Workspace carve (~52.4 MB): xb | WqkvT | WoT | proj | merged  (all bf16)
// ---------------------------------------------------------------------------
extern "C" void kernel_launch(void* const* d_in, const int* in_sizes, int n_in,
                              void* d_out, int out_size, void* d_ws, size_t ws_size,
                              hipStream_t stream) {
  const float* x    = (const float*)d_in[0];
  const float* Wqkv = (const float*)d_in[1];
  const float* bqkv = (const float*)d_in[2];
  const float* Wo   = (const float*)d_in[3];
  const float* bo   = (const float*)d_in[4];
  float* out = (float*)d_out;

  char* w = (char*)d_ws;
  auto carve = [&](size_t bytes) {
    char* p = w;
    w += (bytes + 255) & ~(size_t)255;
    return p;
  };
  __bf16* xb      = (__bf16*)carve((size_t)MROWS * DM * 2);   // 16 MB
  __bf16* WqkvT   = (__bf16*)carve((size_t)DM * DM * 2);      //  2 MB
  __bf16* WoT     = (__bf16*)carve((size_t)DM * DM * 2);      //  2 MB
  __bf16* projb   = (__bf16*)carve((size_t)MROWS * DM * 2);   // 16 MB
  __bf16* mergedb = (__bf16*)carve((size_t)MROWS * DM * 2);   // 16 MB

  const int nX = MROWS * DM;   // 8388608
  cvt_f32_to_bf16<<<nX / 4 / 256, 256, 0, stream>>>(x, xb, nX);
  cvt_transpose_bf16<<<(DM * DM) / 256, 256, 0, stream>>>(Wqkv, WqkvT, DM);
  cvt_transpose_bf16<<<(DM * DM) / 256, 256, 0, stream>>>(Wo, WoT, DM);

  // proj = x @ Wqkv + bqkv   (bf16 out)
  gemm_wmma_bf16<true><<<dim3(MROWS / TM, DM / TN), 256, 0, stream>>>(
      xb, WqkvT, bqkv, nullptr, projb, MROWS, DM, DM);

  // flash attention over all (b,h,qTile)
  attn_wmma<<<dim3(SEQ / 64, HEADS, BATCH), 128, 0, stream>>>(projb, mergedb);

  // out = merged @ Wo + bo   (fp32 out)
  gemm_wmma_bf16<false><<<dim3(MROWS / TM, DM / TN), 256, 0, stream>>>(
      mergedb, WoT, bo, out, nullptr, MROWS, DM, DM);
}
